// AE_14413910245386
// MI455X (gfx1250) — compile-verified, matrix operand order checked by reading, not run on hardware
//
#include <hip/hip_runtime.h>
#include <hip/hip_bf16.h>
#include <math.h>

typedef __attribute__((ext_vector_type(16))) _Float16 v16h;
typedef __attribute__((ext_vector_type(8)))  _Float16 v8h;
typedef __attribute__((ext_vector_type(8)))  float    v8f;

#define NROWS 131072
#define DIM   224
#define HID   400
#define ZDIM  4
#define HP    416   // HID padded to multiple of 32
#define ZP    32    // ZDIM padded to one K-chunk
#define TOPK  20
#define BR    128   // rows per block (8 stripes of 16)

// workspace layout (in f16 elements for the weight region)
#define WE1T_OFF 0                          // [400][224]
#define WE2T_OFF (WE1T_OFF + 400*224)       // [400][416]
#define WE3T_OFF (WE2T_OFF + 400*416)       // [16][416]
#define WD1T_OFF (WE3T_OFF + 16*416)        // [400][32]
#define WD2T_OFF (WD1T_OFF + 400*32)        // [400][416]
#define WD3T_OFF (WD2T_OFF + 400*416)       // [224][416]
#define WT_HALVES (WD3T_OFF + 224*416)

// ---------------- fragment loaders (CDNA5 16-bit WMMA layouts) ----------------
__device__ __forceinline__ v16h frag_a(const _Float16* buf, int stride, int k0, int lane) {
    int row = lane & 15;
    int kb  = (lane & 16) ? 8 : 0;
    const _Float16* p = buf + row * stride + k0 + kb;
    v8h lo = *(const v8h*)p;
    v8h hi = *(const v8h*)(p + 16);
    v16h a;
#pragma unroll
    for (int i = 0; i < 8; ++i) { a[i] = lo[i]; a[8 + i] = hi[i]; }
    return a;
}

__device__ __forceinline__ v16h frag_b(const _Float16* Wt, int Kp, int col, int k0, int lane) {
    int kb = (lane & 16) ? 8 : 0;
    const _Float16* p = Wt + (size_t)col * Kp + k0 + kb;
    v8h lo = *(const v8h*)p;
    v8h hi = *(const v8h*)(p + 16);
    v16h b;
#pragma unroll
    for (int i = 0; i < 8; ++i) { b[i] = lo[i]; b[8 + i] = hi[i]; }
    return b;
}

// one lrelu GEMM layer over 8 row-stripes; N-tiles split across the 8 waves.
// B fragment loaded once per (nt,kc) and reused by 8 WMMAs (register reuse).
// k-loop kept ROLLED so A fragments cannot be hoisted across nt (avoids spills).
__device__ __forceinline__ void layer_ms(const _Float16* inbuf, int inStride, int KC,
                                         const _Float16* Wt, int Kp,
                                         const float* bias, int NT,
                                         _Float16* outbuf, int outStride,
                                         int wave, int lane) {
    int colOff = lane & 15;
    int rowOff = (lane & 16) ? 8 : 0;
    for (int nt = wave; nt < NT; nt += 8) {
        int col = nt * 16 + colOff;
        __builtin_prefetch(Wt + (size_t)col * Kp, 0, 0);
        float bv = bias[col];
        v8f acc[8];
#pragma unroll
        for (int s = 0; s < 8; ++s)
#pragma unroll
            for (int r = 0; r < 8; ++r) acc[s][r] = bv;
#pragma clang loop unroll(disable)
        for (int kc = 0; kc < KC; ++kc) {
            v16h w = frag_b(Wt, Kp, col, kc * 32, lane);
            v16h a[8];
#pragma unroll
            for (int s = 0; s < 8; ++s)
                a[s] = frag_a(inbuf + s * 16 * inStride, inStride, kc * 32, lane);
#pragma unroll
            for (int s = 0; s < 8; ++s)
                acc[s] = __builtin_amdgcn_wmma_f32_16x16x32_f16(false, a[s], false, w,
                                                               (short)0, acc[s], false, false);
        }
#pragma unroll
        for (int s = 0; s < 8; ++s)
#pragma unroll
            for (int r = 0; r < 8; ++r) {
                float v = acc[s][r];
                v = (v > 0.f) ? v : 0.1f * v;
                outbuf[(s * 16 + r + rowOff) * outStride + col] = (_Float16)v;
            }
    }
}

// ---------------- prep: transpose+pad weights to f16, ||d||, zero mse ----------------
__device__ __forceinline__ void transpose_pad(const float* W, int K, int Nn,
                                              _Float16* out, int Kp, int Np,
                                              int tid, int nthreads) {
    int total = Np * Kp;
    for (int i = tid; i < total; i += nthreads) {
        int n = i / Kp, k = i % Kp;
        float v = (n < Nn && k < K) ? W[(size_t)k * Nn + n] : 0.f;
        out[(size_t)n * Kp + k] = (_Float16)v;
    }
}

__global__ void ae_prep(const float* We1, const float* We2, const float* We3,
                        const float* Wd1, const float* Wd2, const float* Wd3,
                        const float* dvec, _Float16* wbuf, float* scal) {
    int tid = blockIdx.x * blockDim.x + threadIdx.x;
    int nt  = gridDim.x * blockDim.x;
    transpose_pad(We1, DIM, HID, wbuf + WE1T_OFF, DIM, HID, tid, nt);
    transpose_pad(We2, HID, HID, wbuf + WE2T_OFF, HP,  HID, tid, nt);
    transpose_pad(We3, HID, ZDIM, wbuf + WE3T_OFF, HP, 16,  tid, nt);
    transpose_pad(Wd1, ZDIM, HID, wbuf + WD1T_OFF, ZP, HID, tid, nt);
    transpose_pad(Wd2, HID, HID, wbuf + WD2T_OFF, HP,  HID, tid, nt);
    transpose_pad(Wd3, HID, DIM, wbuf + WD3T_OFF, HP,  DIM, tid, nt);
    if (tid == 0) {
        float c = 0.f;
        for (int i = 0; i < DIM; ++i) c += dvec[i] * dvec[i];
        scal[0] = 0.f;        // mse accumulator
        scal[1] = sqrtf(c);   // ||d||
    }
}

// ---------------- fused 6-layer autoencoder, 128 rows / block (8 waves) ----------------
__global__ __launch_bounds__(256) void ae_fused(
    const float* __restrict__ x, const float* __restrict__ dvec,
    const float* __restrict__ be1, const float* __restrict__ be2, const float* __restrict__ be3,
    const float* __restrict__ bd1, const float* __restrict__ bd2, const float* __restrict__ bd3,
    const _Float16* __restrict__ wbuf,
    float* __restrict__ scal, float* __restrict__ defen,
    float* __restrict__ out_y, float* __restrict__ out_z) {
    __shared__ __align__(16) _Float16 buf0[BR * HP];   // 104 KB
    __shared__ __align__(16) _Float16 buf1[BR * HP];   // 104 KB
    __shared__ float s_ydot[BR];
    __shared__ float s_ynorm[BR];

    int tid  = threadIdx.x;
    int wave = tid >> 5;
    int lane = tid & 31;
    int rowBase = blockIdx.x * BR;

    // zero both LDS buffers (establishes all pad regions) + row accumulators
    {
        v8h zz = {};
        for (int i = tid; i < (BR * HP) / 8; i += 256) {
            ((v8h*)buf0)[i] = zz;
            ((v8h*)buf1)[i] = zz;
        }
        if (tid < BR) { s_ydot[tid] = 0.f; s_ynorm[tid] = 0.f; }
    }
    __syncthreads();
    // stage x -> buf0 as f16 (stride DIM = 224 = 7*32): wave w stages rows 16w..16w+15
    for (int r = 0; r < 16; ++r) {
        int lrow = wave * 16 + r;
        const float* xr = x + (size_t)(rowBase + lrow) * DIM;
        for (int c = lane; c < DIM; c += 32)
            buf0[lrow * DIM + c] = (_Float16)xr[c];
    }
    __syncthreads();

    // L1: h1 = lrelu(x @ We1 + be1)          buf0 -> buf1
    layer_ms(buf0, DIM, DIM / 32, wbuf + WE1T_OFF, DIM, be1, HID / 16, buf1, HP, wave, lane);
    __syncthreads();
    // re-zero pad columns of buf0 (it held x at a different stride)
    for (int i = tid; i < BR * 16; i += 256)
        buf0[(i >> 4) * HP + HID + (i & 15)] = (_Float16)0.f;
    __syncthreads();
    // L2: h2 = lrelu(h1 @ We2 + be2)         buf1 -> buf0
    layer_ms(buf1, HP, HP / 32, wbuf + WE2T_OFF, HP, be2, HID / 16, buf0, HP, wave, lane);
    __syncthreads();

    // L3: z = h2 @ We3 + be3 (one 16-col tile, 4 valid); wave w does stripe w
    {
        int s = wave;
        int colOff = lane & 15;
        int rowOff = (lane & 16) ? 8 : 0;
        float bv = (colOff < ZDIM) ? be3[colOff] : 0.f;
        v8f acc;
#pragma unroll
        for (int r = 0; r < 8; ++r) acc[r] = bv;
#pragma clang loop unroll(disable)
        for (int kc = 0; kc < HP / 32; ++kc) {
            v16h a = frag_a(buf0 + s * 16 * HP, HP, kc * 32, lane);
            v16h w = frag_b(wbuf + WE3T_OFF, HP, colOff, kc * 32, lane);
            acc = __builtin_amdgcn_wmma_f32_16x16x32_f16(false, a, false, w,
                                                         (short)0, acc, false, false);
        }
#pragma unroll
        for (int r = 0; r < 8; ++r) {
            float v = acc[r];
            int lrow = s * 16 + r + rowOff;
            if (colOff < ZDIM)
                out_z[(size_t)(rowBase + lrow) * ZDIM + colOff] = v;
            buf1[lrow * ZP + colOff] = (_Float16)v;   // cols 4..15 are exactly 0
        }
        for (int i = lane; i < 256; i += 32)          // zero cols 16..31 of this stripe
            buf1[(s * 16 + (i >> 4)) * ZP + 16 + (i & 15)] = (_Float16)0.f;
    }
    __syncthreads();

    // L4: g1 = lrelu(z @ Wd1 + bd1)          buf1 -> buf0 (K = one 32 chunk)
    layer_ms(buf1, ZP, 1, wbuf + WD1T_OFF, ZP, bd1, HID / 16, buf0, HP, wave, lane);
    __syncthreads();
    // re-zero pad columns of buf1 (z layout disturbed the low region)
    for (int i = tid; i < BR * 16; i += 256)
        buf1[(i >> 4) * HP + HID + (i & 15)] = (_Float16)0.f;
    __syncthreads();
    // L5: g2 = lrelu(g1 @ Wd2 + bd2)         buf0 -> buf1
    layer_ms(buf0, HP, HP / 32, wbuf + WD2T_OFF, HP, bd2, HID / 16, buf1, HP, wave, lane);
    __syncthreads();

    // L6: y = tanh(g2 @ Wd3 + bd3) fused with y store, y.d, ||y||, mse partials
    {
        int colOff = lane & 15;
        int rowOff = (lane & 16) ? 8 : 0;
        int lid    = lane & 15;
        float msum = 0.f;
        for (int nt = wave; nt < DIM / 16; nt += 8) {
            int col = nt * 16 + colOff;
            float bv = bd3[col];
            v8f acc[8];
#pragma unroll
            for (int s = 0; s < 8; ++s)
#pragma unroll
                for (int r = 0; r < 8; ++r) acc[s][r] = bv;
#pragma clang loop unroll(disable)
            for (int kc = 0; kc < HP / 32; ++kc) {
                v16h w = frag_b(wbuf + WD3T_OFF, HP, col, kc * 32, lane);
                v16h a[8];
#pragma unroll
                for (int s = 0; s < 8; ++s)
                    a[s] = frag_a(buf1 + s * 16 * HP, HP, kc * 32, lane);
#pragma unroll
                for (int s = 0; s < 8; ++s)
                    acc[s] = __builtin_amdgcn_wmma_f32_16x16x32_f16(false, a[s], false, w,
                                                                   (short)0, acc[s], false, false);
            }
            float dval = dvec[col];
#pragma unroll
            for (int s = 0; s < 8; ++s) {
                float yd[8], yn[8];
#pragma unroll
                for (int r = 0; r < 8; ++r) {
                    int row = rowBase + s * 16 + r + rowOff;
                    float v = tanhf(acc[s][r]);
                    size_t idx = (size_t)row * DIM + col;
                    out_y[idx] = v;
                    float e = v - x[idx];
                    msum += e * e;
                    yd[r] = v * dval;
                    yn[r] = v * v;
                }
                // reduce this stripe's row partials across each 16-lane half-group
#pragma unroll
                for (int off = 1; off < 16; off <<= 1) {
#pragma unroll
                    for (int r = 0; r < 8; ++r) {
                        yd[r] += __shfl_xor(yd[r], off, 16);
                        yn[r] += __shfl_xor(yn[r], off, 16);
                    }
                }
#pragma unroll
                for (int r = 0; r < 8; ++r) {
                    if (lid == r) {
                        int lrow = s * 16 + r + rowOff;
                        atomicAdd(&s_ydot[lrow],  yd[r]);
                        atomicAdd(&s_ynorm[lrow], yn[r]);
                    }
                }
            }
        }
        // full-wave reduction of mse partial
#pragma unroll
        for (int off = 1; off < 32; off <<= 1)
            msum += __shfl_xor(msum, off, 32);
        if (lane == 0) atomicAdd(&scal[0], msum);
    }
    __syncthreads();
    if (tid < BR) {
        float Cn = scal[1];
        float A  = s_ydot[tid];
        float Bn = sqrtf(s_ynorm[tid]);
        defen[rowBase + tid] = A / (Bn * Cn + 1e-5f);
    }
}

// ---------------- exact top-20 + final scalar ----------------
__global__ __launch_bounds__(256) void ae_topk(const float* __restrict__ defen,
                                               const float* __restrict__ scal,
                                               float* __restrict__ out_r) {
    __shared__ float cand[256 * TOPK];
    __shared__ float sval[256];
    __shared__ int   sidx[256];
    int t = threadIdx.x;
    float top[TOPK];
#pragma unroll
    for (int i = 0; i < TOPK; ++i) top[i] = -INFINITY;
    for (int i = t; i < NROWS; i += 256) {
        float v = defen[i];
        if (v > top[TOPK - 1]) {
            int j = TOPK - 1;
            while (j > 0 && v > top[j - 1]) { top[j] = top[j - 1]; --j; }
            top[j] = v;
        }
    }
#pragma unroll
    for (int i = 0; i < TOPK; ++i) cand[t * TOPK + i] = top[i];
    __syncthreads();
    float ssum = 0.f;
    for (int round = 0; round < TOPK; ++round) {
        float bv = -INFINITY; int bi = 0;
        for (int i = t; i < 256 * TOPK; i += 256) {
            float v = cand[i];
            if (v > bv) { bv = v; bi = i; }
        }
        sval[t] = bv; sidx[t] = bi;
        __syncthreads();
        for (int s = 128; s > 0; s >>= 1) {
            if (t < s && sval[t + s] > sval[t]) { sval[t] = sval[t + s]; sidx[t] = sidx[t + s]; }
            __syncthreads();
        }
        if (t == 0) { ssum += sval[0]; cand[sidx[0]] = -INFINITY; }
        __syncthreads();
    }
    if (t == 0) {
        float mse = scal[0] / (float)((size_t)NROWS * DIM);
        out_r[0] = mse + 0.1f * ssum;
    }
}

extern "C" void kernel_launch(void* const* d_in, const int* in_sizes, int n_in,
                              void* d_out, int out_size, void* d_ws, size_t ws_size,
                              hipStream_t stream) {
    (void)in_sizes; (void)n_in; (void)out_size; (void)ws_size;
    const float* x    = (const float*)d_in[0];
    const float* dvec = (const float*)d_in[1];
    const float* We1  = (const float*)d_in[2];
    const float* be1  = (const float*)d_in[3];
    const float* We2  = (const float*)d_in[4];
    const float* be2  = (const float*)d_in[5];
    const float* We3  = (const float*)d_in[6];
    const float* be3  = (const float*)d_in[7];
    const float* Wd1  = (const float*)d_in[8];
    const float* bd1  = (const float*)d_in[9];
    const float* Wd2  = (const float*)d_in[10];
    const float* bd2  = (const float*)d_in[11];
    const float* Wd3  = (const float*)d_in[12];
    const float* bd3  = (const float*)d_in[13];

    float* out   = (float*)d_out;
    float* out_y = out;
    float* out_z = out + (size_t)NROWS * DIM;
    float* out_r = out + (size_t)NROWS * DIM + (size_t)NROWS * ZDIM;

    _Float16* wbuf = (_Float16*)d_ws;
    size_t wbytes  = ((size_t)WT_HALVES * 2 + 255) & ~(size_t)255;
    float* defen   = (float*)((char*)d_ws + wbytes);
    float* scal    = defen + NROWS;   // [0] = mse accum, [1] = ||d||

    ae_prep<<<64, 256, 0, stream>>>(We1, We2, We3, Wd1, Wd2, Wd3, dvec, wbuf, scal);
    ae_fused<<<NROWS / BR, 256, 0, stream>>>(x, dvec, be1, be2, be3, bd1, bd2, bd3,
                                             wbuf, scal, defen, out_y, out_z);
    ae_topk<<<1, 256, 0, stream>>>(defen, scal, out_r);
}
